// GraphSAGE_10273561772524
// MI455X (gfx1250) — matrix operand order repaired and checked by reading, not compiled, not run on hardware
//
#include <hip/hip_runtime.h>
#include <stdint.h>

typedef __attribute__((ext_vector_type(16))) __bf16 v16bf;
typedef __attribute__((ext_vector_type(8)))  float  v8f;

union BF4    { __bf16 b[4]; uint2 u; };          // 4 bf16 <-> 8 bytes
union Frag16 { uint4 q[2];  v16bf v; };          // 16 bf16 <-> 2 x b128

// ---------------------------------------------------------------------------
// Pre-pack weights: W[K,NF] f32 -> bf16 in WMMA B-fragment lane order.
// Packed dword index = ((kchunk*(NF/16) + ntile)*32 + lane)*8 + p,
// dword p holds (K = kchunk*32 + hi*16 + 2p [+1 in high half]) for col
// ntile*16 + (lane&15).  One wave-chunk = 1KB contiguous -> b128 loads.
// k-chunk-major => two matrices packed back-to-back form a fused [W0;W1].
// ---------------------------------------------------------------------------
template<int K, int NF>
__global__ void pack_w(const float* __restrict__ W, uint32_t* __restrict__ Wp)
{
    const int idx = blockIdx.x * blockDim.x + threadIdx.x;
    if (idx >= K * NF / 2) return;
    const int p    = idx & 7;
    const int lane = (idx >> 3) & 31;
    const int t    = idx >> 8;                 // kchunk*(NF/16) + ntile
    constexpr int NT = NF / 16;
    const int kchunk = t / NT, nt = t % NT;
    const int col = nt * 16 + (lane & 15);
    const int k0  = kchunk * 32 + (lane >> 4) * 16 + p * 2;
    union { __bf16 b[2]; uint32_t u; } pk;
    pk.b[0] = (__bf16)W[(size_t)k0 * NF + col];
    pk.b[1] = (__bf16)W[(size_t)(k0 + 1) * NF + col];
    Wp[idx] = pk.u;
}

// ---------------------------------------------------------------------------
// Fused WMMA GEMM: out[M,NF] = ReLU?( [A0|A1][M,K0+K1] @ Wp + bias )
//  - bf16 16x16x32 WMMA, f32 accumulate; MT=4 16-row M tiles per block
//    (4 independent WMMAs per B fragment -> hides D->A/B hazard, 4x B reuse)
//  - A0/A1 staged to one LDS tile as bf16 (row stride K+8 -> 16B-aligned
//    conflict-free b128 fragment loads)
//  - B fragments: 2 x global_load_b128 from pre-packed weights (L2-hot)
//  - K1 == 0 folds to a plain single-source GEMM
// ---------------------------------------------------------------------------
template<int K0, int K1, int NF, int MT, bool RELU>
__global__ void __launch_bounds__(32 * (NF / 16))
sage_gemm(const float* __restrict__ A0, const float* __restrict__ A1,
          const uint32_t* __restrict__ Wp, const float* __restrict__ bias,
          float* __restrict__ out, int M)
{
    constexpr int K    = K0 + K1;
    constexpr int NW   = NF / 16;
    constexpr int PADK = K + 8;                  // row stride stays 16B aligned
    __shared__ __bf16 Alds[MT * 16 * PADK];

    const int tid   = threadIdx.x;
    const int lane  = tid & 31;
    const int wave  = tid >> 5;
    const int mbase = blockIdx.x * (MT * 16);

    // ---- stage [A0|A1] (MT*16 x K) into LDS as bf16 ----
    constexpr int NV = MT * 16 * K / 4;
    for (int idx = tid; idx < NV; idx += 32 * NW) {
        const int r  = idx / (K / 4);
        const int cc = (idx % (K / 4)) * 4;
        const int row = mbase + r;
        float4 v = make_float4(0.f, 0.f, 0.f, 0.f);
        if (row < M) {
            if (K1 == 0 || cc < K0)
                v = *(const float4*)(A0 + (size_t)row * K0 + cc);
            else
                v = *(const float4*)(A1 + (size_t)row * K1 + (cc - K0));
        }
        BF4 pk;
        pk.b[0] = (__bf16)v.x; pk.b[1] = (__bf16)v.y;
        pk.b[2] = (__bf16)v.z; pk.b[3] = (__bf16)v.w;
        *(uint2*)&Alds[r * PADK + cc] = pk.u;
    }
    __syncthreads();

    const int mrow = lane & 15;
    const int hi   = lane >> 4;
    const int ncol = wave * 16 + (lane & 15);

    // ---- accumulators: bias splat ----
    v8f acc[MT];
    {
        const float bv = bias[ncol];
#pragma unroll
        for (int mt = 0; mt < MT; ++mt)
#pragma unroll
            for (int j = 0; j < 8; ++j) acc[mt][j] = bv;
    }

    // ---- K loop: load B frag once, MT independent WMMAs ----
#pragma unroll
    for (int kk = 0; kk < K; kk += 32) {
        Frag16 bf;
        const uint4* wp4 = (const uint4*)Wp +
                           ((size_t)(kk / 32) * NW + wave) * 64 + lane * 2;
        bf.q[0] = wp4[0];
        bf.q[1] = wp4[1];
#pragma unroll
        for (int mt = 0; mt < MT; ++mt) {
            Frag16 af;
            const __bf16* ab = &Alds[(mt * 16 + mrow) * PADK + kk + hi * 16];
            af.q[0] = *(const uint4*)ab;            // K 0..7   (+hi*16)
            af.q[1] = *(const uint4*)(ab + 8);      // K 8..15  (+hi*16)
            acc[mt] = __builtin_amdgcn_wmma_f32_16x16x32_bf16(
                false, af.v, false, bf.v, (short)0, acc[mt], false, false);
        }
    }

    // ---- epilogue: ReLU must emit +0 (uint-max segment trick needs it) ----
#pragma unroll
    for (int mt = 0; mt < MT; ++mt) {
#pragma unroll
        for (int j = 0; j < 8; ++j) {
            const int row = mbase + mt * 16 + j + hi * 8;
            if (row < M) {
                float v = acc[mt][j];
                if (RELU) v = (v > 0.0f) ? v : 0.0f;
                out[(size_t)row * NF + ncol] = v;
            }
        }
    }
}

// ---------------------------------------------------------------------------
// Edge scatter-max: agg[dst] = max(agg[dst], pooled[src]) per feature.
// pooled >= 0 (post-ReLU, +0 only) => uint bit-pattern atomicMax == float max.
// Thread handles one edge x 4 features (float4 gather, 4 uint atomics).
// pooled/agg are L2-resident (<= 51MB each vs 192MB L2).
// ---------------------------------------------------------------------------
template<int F>
__global__ void edge_segment_max(const float* __restrict__ pooled,
                                 const int* __restrict__ src,
                                 const int* __restrict__ dst,
                                 unsigned int* __restrict__ agg, int E)
{
    constexpr int CH = F / 4;
    const int total  = E * CH;
    for (int tid = blockIdx.x * blockDim.x + threadIdx.x; tid < total;
         tid += gridDim.x * blockDim.x) {
        const int e = tid / CH;
        const int c = (tid % CH) * 4;
        const int s = src[e], d = dst[e];
        const float4 v = *(const float4*)(pooled + (size_t)s * F + c);
        unsigned int* p = agg + (size_t)d * F + c;
        atomicMax(p + 0, __float_as_uint(v.x));
        atomicMax(p + 1, __float_as_uint(v.y));
        atomicMax(p + 2, __float_as_uint(v.z));
        atomicMax(p + 3, __float_as_uint(v.w));
    }
}

// ---------------------------------------------------------------------------
// Output linear: [N,64] @ [64,2] + b  (tiny, VALU)
// ---------------------------------------------------------------------------
__global__ void out_linear(const float* __restrict__ h,
                           const float* __restrict__ Wo,
                           const float* __restrict__ bo,
                           float* __restrict__ out, int M)
{
    const int n = blockIdx.x * blockDim.x + threadIdx.x;
    if (n >= M) return;
    float a0 = bo[0], a1 = bo[1];
    const float* hr = h + (size_t)n * 64;
#pragma unroll 8
    for (int k = 0; k < 64; ++k) {
        const float v = hr[k];
        a0 += v * Wo[k * 2 + 0];
        a1 += v * Wo[k * 2 + 1];
    }
    out[(size_t)n * 2 + 0] = a0;
    out[(size_t)n * 2 + 1] = a1;
}

extern "C" void kernel_launch(void* const* d_in, const int* in_sizes, int n_in,
                              void* d_out, int out_size, void* d_ws, size_t ws_size,
                              hipStream_t stream)
{
    const float* x        = (const float*)d_in[0];
    const int*   src      = (const int*)  d_in[1];
    const int*   dst      = (const int*)  d_in[2];
    const float* W_pool1  = (const float*)d_in[3];
    const float* b_pool1  = (const float*)d_in[4];
    const float* W_self1  = (const float*)d_in[5];
    const float* W_neigh1 = (const float*)d_in[6];
    const float* b1       = (const float*)d_in[7];
    const float* W_pool2  = (const float*)d_in[8];
    const float* b_pool2  = (const float*)d_in[9];
    const float* W_self2  = (const float*)d_in[10];
    const float* W_neigh2 = (const float*)d_in[11];
    const float* b2       = (const float*)d_in[12];
    const float* W_out    = (const float*)d_in[13];
    const float* b_out    = (const float*)d_in[14];

    const int N = in_sizes[0] / 128;   // 100000
    const int E = in_sizes[1];         // 1600000

    // Workspace (floats): activations N*256 (~102MB, buffers recycled),
    // then 22528 dwords of packed bf16 weights (~90KB).
    float* ws      = (float*)d_ws;
    float* pooled1 = ws;                        // [0, N*128)
    float* agg1    = ws + (size_t)N * 128;      // [N*128, N*256)
    float* h1      = ws;                        // reuse pooled1 (dead)
    float* pooled2 = ws + (size_t)N * 64;
    float* agg2    = ws + (size_t)N * 128;      // reuse agg1 (dead)
    float* h2      = ws + (size_t)N * 192;
    uint32_t* wpk      = (uint32_t*)(ws + (size_t)N * 256);
    uint32_t* wp_pool1 = wpk;                   // 8192 dwords (128x128)
    uint32_t* wp_sn1   = wpk + 8192;            // 8192: [W_self1;W_neigh1] K=256
    uint32_t* wp_pool2 = wpk + 16384;           // 2048 (64x64)
    uint32_t* wp_sn2   = wpk + 18432;           // 4096: [W_self2;W_neigh2] K=128
    if (ws_size < ((size_t)N * 256 + 22528) * sizeof(float)) return;

    // ---- pack all weights to bf16 WMMA-B layout (tiny) ----
    // Fused buffers: per-kchunk block is (NF/16)*256 dwords, so the second
    // matrix starts at (K0/32)*(NF/16)*256 dwords into the fused buffer.
    pack_w<128, 128><<<(8192 + 255) / 256, 256, 0, stream>>>(W_pool1,  wp_pool1);
    pack_w<128, 64 ><<<(4096 + 255) / 256, 256, 0, stream>>>(W_self1,  wp_sn1);
    pack_w<128, 64 ><<<(4096 + 255) / 256, 256, 0, stream>>>(W_neigh1, wp_sn1 + 4096);
    pack_w<64,  64 ><<<(2048 + 255) / 256, 256, 0, stream>>>(W_pool2,  wp_pool2);
    pack_w<64,  64 ><<<(2048 + 255) / 256, 256, 0, stream>>>(W_self2,  wp_sn2);
    pack_w<64,  64 ><<<(2048 + 255) / 256, 256, 0, stream>>>(W_neigh2, wp_sn2 + 2048);

    const int mtiles = (N + 63) / 64;  // MT=4 -> 1563 blocks

    // ---------------- layer 1 (128 -> 64) ----------------
    sage_gemm<128, 0, 128, 4, true><<<mtiles, 256, 0, stream>>>(
        x, nullptr, wp_pool1, b_pool1, pooled1, N);

    hipMemsetAsync(agg1, 0, (size_t)N * 128 * sizeof(float), stream);
    {
        const int total  = E * (128 / 4);
        const int blocks = (total + 255) / 256;
        edge_segment_max<128><<<blocks, 256, 0, stream>>>(
            pooled1, src, dst, (unsigned int*)agg1, E);
    }
    // h1 = ReLU([x | agg1] @ [W_self1; W_neigh1] + b1)   (fused, K=256)
    sage_gemm<128, 128, 64, 4, true><<<mtiles, 128, 0, stream>>>(
        x, agg1, wp_sn1, b1, h1, N);

    // ---------------- layer 2 (64 -> 64) ----------------
    sage_gemm<64, 0, 64, 4, true><<<mtiles, 128, 0, stream>>>(
        h1, nullptr, wp_pool2, b_pool2, pooled2, N);

    hipMemsetAsync(agg2, 0, (size_t)N * 64 * sizeof(float), stream);
    {
        const int total  = E * (64 / 4);
        const int blocks = (total + 255) / 256;
        edge_segment_max<64><<<blocks, 256, 0, stream>>>(
            pooled2, src, dst, (unsigned int*)agg2, E);
    }
    // h2 = ReLU([h1 | agg2] @ [W_self2; W_neigh2] + b2)   (fused, K=128)
    sage_gemm<64, 64, 64, 4, true><<<mtiles, 128, 0, stream>>>(
        h1, agg2, wp_sn2, b2, h2, N);

    // ---------------- output linear ----------------
    out_linear<<<(N + 255) / 256, 256, 0, stream>>>(
        h2, W_out, b_out, (float*)d_out, N);
}
